// Attention_46514495816362
// MI455X (gfx1250) — matrix operand order repaired and checked by reading, not compiled, not run on hardware
//
#include <hip/hip_runtime.h>
#include <hip/hip_bf16.h>

// ---------------------------------------------------------------------------
// Problem constants (from the reference)
// ---------------------------------------------------------------------------
#define NTOK 1168            // N = 12*12 + 32*32  (73 * 16)
#define NPAD 1184            // padded to multiple of 32 for 32-key flash blocks
#define DIMC 256
#define NH   8
#define DH   32
#define BB   16
#define MTOT (BB * NTOK)     // 18688 rows (= 584 * 32 = 1168 * 16)
#define QK_SCALE 0.17677669529663687f   // 32^-0.5

typedef __attribute__((ext_vector_type(16))) __bf16 v16bf;
typedef __attribute__((ext_vector_type(8)))  float  v8f;

union FragBF {
  v16bf v;
  unsigned short s[16];
  uint4 u[2];
};
union Acc8 {
  v8f   v;
  float f[8];
};

static __device__ inline v8f vzero8() {
  v8f z = {0.f, 0.f, 0.f, 0.f, 0.f, 0.f, 0.f, 0.f};
  return z;
}

// round-to-nearest-even f32 -> bf16 bits
static __device__ inline unsigned short f2bf(float x) {
  union { float f; unsigned u; } c;
  c.f = x;
  unsigned u = c.u + 0x7FFFu + ((c.u >> 16) & 1u);
  return (unsigned short)(u >> 16);
}

// ---------------------------------------------------------------------------
// fp32 -> bf16 conversion
// ---------------------------------------------------------------------------
__global__ void cvt_f32_bf16(const float* __restrict__ src,
                             unsigned short* __restrict__ dst, int n) {
  int i = blockIdx.x * blockDim.x + threadIdx.x;
  if (i < n) dst[i] = f2bf(src[i]);
}

// ---------------------------------------------------------------------------
// QKV GEMM: C[m,n] = sum_k X[m,k] * Wqkv[n,k], m in [0,18688), n in [0,768)
// One wave computes a 32x64 register tile: 2 A-frags x 4 B-frags -> 8 WMMAs
// per 32-wide K step (1.5 b128 loads per WMMA).
// Epilogue scatters into attention-friendly layouts:
//   Q: [B,H,NPAD,32] (scaled), K: [B,H,NPAD,32], V^T: [B,H,32,NPAD]
// ---------------------------------------------------------------------------
__global__ void qkv_gemm(const unsigned short* __restrict__ Xb,
                         const unsigned short* __restrict__ Wb,
                         unsigned short* __restrict__ Qb,
                         unsigned short* __restrict__ Kb,
                         unsigned short* __restrict__ Vtb) {
  const int lane = threadIdx.x;       // 0..31 (wave32)
  const int c16  = lane & 15;
  const int half = lane >> 4;
  const int m0 = blockIdx.x * 32;     // 584 tiles
  const int n0 = blockIdx.y * 64;     // 12 tiles

  Acc8 acc[2][4];
  #pragma unroll
  for (int i = 0; i < 2; ++i)
    #pragma unroll
    for (int j = 0; j < 4; ++j) acc[i][j].v = vzero8();

  const unsigned short* arow[2];
  arow[0] = Xb + (size_t)(m0 + c16) * DIMC;
  arow[1] = Xb + (size_t)(m0 + 16 + c16) * DIMC;
  const unsigned short* brow[4];
  #pragma unroll
  for (int j = 0; j < 4; ++j)
    brow[j] = Wb + (size_t)(n0 + j * 16 + c16) * DIMC;

  #pragma unroll
  for (int k0 = 0; k0 < DIMC; k0 += 32) {
    FragBF a[2], b[4];
    #pragma unroll
    for (int i = 0; i < 2; ++i) {
      // A 16x32 bf16 fragment: lane row = c16; elems 0..7 -> K=k0+half*8+e,
      // elems 8..15 -> K=k0+16+half*8+(e-8)
      a[i].u[0] = *(const uint4*)(arow[i] + k0 + half * 8);
      a[i].u[1] = *(const uint4*)(arow[i] + k0 + 16 + half * 8);
    }
    #pragma unroll
    for (int j = 0; j < 4; ++j) {
      // B 32x16 bf16 fragment: lane col = c16; elems 0..7 -> K=k0+half*8+e,
      // elems 8..15 -> K=k0+16+half*8+(e-8)
      b[j].u[0] = *(const uint4*)(brow[j] + k0 + half * 8);
      b[j].u[1] = *(const uint4*)(brow[j] + k0 + 16 + half * 8);
    }
    #pragma unroll
    for (int i = 0; i < 2; ++i)
      #pragma unroll
      for (int j = 0; j < 4; ++j)
        acc[i][j].v = __builtin_amdgcn_wmma_f32_16x16x32_bf16(
            false, a[i].v, false, b[j].v, (short)0, acc[i][j].v, false, false);
  }

  // epilogue: each 16-row subtile is fully inside one batch (1168 = 73*16)
  #pragma unroll
  for (int i = 0; i < 2; ++i) {
    const int msub    = m0 + i * 16;
    const int b_idx   = msub / NTOK;
    const int tokbase = msub % NTOK;
    #pragma unroll
    for (int j = 0; j < 4; ++j) {
      const int n = n0 + j * 16 + c16;  // lane's output column
      const int s = n >> 8;             // 0=q, 1=k, 2=v
      const int h = (n >> 5) & 7;
      const int d = n & 31;
      const size_t bh = (size_t)b_idx * NH + h;
      #pragma unroll
      for (int r = 0; r < 8; ++r) {
        int tok = tokbase + r + half * 8;   // C layout: row = r (+8 hi half)
        float val = acc[i][j].f[r];
        if (s == 0) {
          Qb[(bh * NPAD + tok) * DH + d] = f2bf(val * QK_SCALE);
        } else if (s == 1) {
          Kb[(bh * NPAD + tok) * DH + d] = f2bf(val);
        } else {
          Vtb[(bh * DH + d) * NPAD + tok] = f2bf(val);
        }
      }
    }
  }
}

// ---------------------------------------------------------------------------
// Flash attention, transposed-score trick. One wave = (b,h, 16 queries).
// S^T = K·Q^T  -> softmax stats are per-lane (query = lane&15, split across
// lane pair {q, q+16}; one shfl_xor(16) folds them).
// O^T = V^T · P^T -> P^T B-fragment built with one half-wave swap.
// ---------------------------------------------------------------------------
__global__ void attn_kernel(const unsigned short* __restrict__ Qb,
                            const unsigned short* __restrict__ Kb,
                            const unsigned short* __restrict__ Vtb,
                            const float* __restrict__ rpb,
                            const int* __restrict__ rel,
                            const unsigned char* __restrict__ maskp,
                            unsigned short* __restrict__ AOb,
                            int num_rel) {
  const int lane = threadIdx.x;
  const int q16  = lane & 15;
  const int half = lane >> 4;
  const int bh = blockIdx.x;        // 0..127
  const int b  = bh >> 3;
  const int h  = bh & 7;
  const int qtok = blockIdx.y * 16 + q16;

  // Q^T B-fragment: lane col = query q16; elems e -> d = half*8 + e (+16 for e>=8)
  FragBF qf;
  const unsigned short* qrow = Qb + ((size_t)bh * NPAD + qtok) * DH;
  qf.u[0] = *(const uint4*)(qrow + half * 8);
  qf.u[1] = *(const uint4*)(qrow + 16 + half * 8);

  float m_run = -1e30f, l_run = 0.0f;
  Acc8 acc0, acc1;                    // O^T rows d=0..15 and d=16..31
  acc0.v = vzero8();
  acc1.v = vzero8();

  const float* rpbh = rpb + (size_t)h * num_rel;
  const int* relrow = rel + (size_t)qtok * NTOK;
  const unsigned char* maskb = maskp + (size_t)b * NTOK;

  for (int k0 = 0; k0 < NPAD; k0 += 32) {
    // K A-fragments for key tiles [k0,k0+16) and [k0+16,k0+32)
    FragBF ka0, ka1;
    const unsigned short* kr0 = Kb + ((size_t)bh * NPAD + (k0 + q16)) * DH;
    const unsigned short* kr1 = kr0 + 16 * DH;
    ka0.u[0] = *(const uint4*)(kr0 + half * 8);
    ka0.u[1] = *(const uint4*)(kr0 + 16 + half * 8);
    ka1.u[0] = *(const uint4*)(kr1 + half * 8);
    ka1.u[1] = *(const uint4*)(kr1 + 16 + half * 8);

    Acc8 s0, s1;   // S^T tiles: VGPR r -> key = k0(+16) + r + half*8, col = q16
    s0.v = __builtin_amdgcn_wmma_f32_16x16x32_bf16(
        false, ka0.v, false, qf.v, (short)0, vzero8(), false, false);
    s1.v = __builtin_amdgcn_wmma_f32_16x16x32_bf16(
        false, ka1.v, false, qf.v, (short)0, vzero8(), false, false);

    // bias gather + key mask (pad keys >= NTOK are masked)
    float pf0[8], pf1[8];
    float mx = -1e30f;
    #pragma unroll
    for (int r = 0; r < 8; ++r) {
      int key0 = k0 + r + half * 8;
      int key1 = key0 + 16;
      float v0 = -1e30f, v1 = -1e30f;
      if (key0 < NTOK && !maskb[key0]) v0 = s0.f[r] + rpbh[relrow[key0]];
      if (key1 < NTOK && !maskb[key1]) v1 = s1.f[r] + rpbh[relrow[key1]];
      pf0[r] = v0;
      pf1[r] = v1;
      mx = fmaxf(mx, fmaxf(v0, v1));
    }
    mx = fmaxf(mx, __shfl_xor(mx, 16, 32));   // fold lane pair -> column max

    const float mnew  = fmaxf(m_run, mx);
    const float alpha = __expf(m_run - mnew);
    float psum = 0.0f;
    #pragma unroll
    for (int r = 0; r < 8; ++r) {
      float p0 = (pf0[r] < -5e29f) ? 0.0f : __expf(pf0[r] - mnew);
      float p1 = (pf1[r] < -5e29f) ? 0.0f : __expf(pf1[r] - mnew);
      pf0[r] = p0;
      pf1[r] = p1;
      psum += p0 + p1;
    }
    psum += __shfl_xor(psum, 16, 32);
    l_run = l_run * alpha + psum;
    m_run = mnew;
    #pragma unroll
    for (int r = 0; r < 8; ++r) {
      acc0.f[r] *= alpha;
      acc1.f[r] *= alpha;
    }

    // P^T B-fragment (32 keys x 16 queries): lane col=q16, elem e -> key k0+half*16+e.
    // C-layout lane pair swap: keys 8..15 (lo half) / 16..23 (hi half) come from
    // the partner lane via shfl_xor 16.
    FragBF pfrag;
    #pragma unroll
    for (int r = 0; r < 8; ++r) {
      float o0 = __shfl_xor(pf0[r], 16, 32);
      float o1 = __shfl_xor(pf1[r], 16, 32);
      pfrag.s[r]     = f2bf(half == 0 ? pf0[r] : o1);
      pfrag.s[r + 8] = f2bf(half == 0 ? o0 : pf1[r]);
    }

    // V^T A-fragments: rows d (=q16 / 16+q16), elems -> keys k0+half*16+e
    FragBF va0, va1;
    const unsigned short* vr0 = Vtb + ((size_t)bh * DH + q16) * NPAD + k0;
    const unsigned short* vr1 = Vtb + ((size_t)bh * DH + 16 + q16) * NPAD + k0;
    va0.u[0] = *(const uint4*)(vr0 + half * 16);
    va0.u[1] = *(const uint4*)(vr0 + half * 16 + 8);
    va1.u[0] = *(const uint4*)(vr1 + half * 16);
    va1.u[1] = *(const uint4*)(vr1 + half * 16 + 8);

    acc0.v = __builtin_amdgcn_wmma_f32_16x16x32_bf16(
        false, va0.v, false, pfrag.v, (short)0, acc0.v, false, false);
    acc1.v = __builtin_amdgcn_wmma_f32_16x16x32_bf16(
        false, va1.v, false, pfrag.v, (short)0, acc1.v, false, false);
  }

  const float inv = (l_run > 0.0f) ? (1.0f / l_run) : 0.0f;
  // AO[b, tok, h*32 + d] as bf16, row-major [MTOT, 256]
  unsigned short* aorow = AOb + ((size_t)b * NTOK + qtok) * DIMC + h * DH;
  #pragma unroll
  for (int r = 0; r < 8; ++r) {
    int d0 = r + half * 8;            // O^T C layout: row d = r (+8 hi half)
    aorow[d0]      = f2bf(acc0.f[r] * inv);
    aorow[16 + d0] = f2bf(acc1.f[r] * inv);
  }
}

// ---------------------------------------------------------------------------
// Output projection: out[m,n] = sum_k AO[m,k]*Wproj[n,k] + bias[n]  (f32 out)
// 32x64 register tile per wave, same scheme as qkv_gemm.
// ---------------------------------------------------------------------------
__global__ void proj_gemm(const unsigned short* __restrict__ AOb,
                          const unsigned short* __restrict__ Wb,
                          const float* __restrict__ bias,
                          float* __restrict__ out) {
  const int lane = threadIdx.x;
  const int c16  = lane & 15;
  const int half = lane >> 4;
  const int m0 = blockIdx.x * 32;     // 584 tiles
  const int n0 = blockIdx.y * 64;     // 4 tiles

  Acc8 acc[2][4];
  #pragma unroll
  for (int i = 0; i < 2; ++i)
    #pragma unroll
    for (int j = 0; j < 4; ++j) acc[i][j].v = vzero8();

  const unsigned short* arow[2];
  arow[0] = AOb + (size_t)(m0 + c16) * DIMC;
  arow[1] = AOb + (size_t)(m0 + 16 + c16) * DIMC;
  const unsigned short* brow[4];
  #pragma unroll
  for (int j = 0; j < 4; ++j)
    brow[j] = Wb + (size_t)(n0 + j * 16 + c16) * DIMC;

  #pragma unroll
  for (int k0 = 0; k0 < DIMC; k0 += 32) {
    FragBF a[2], b[4];
    #pragma unroll
    for (int i = 0; i < 2; ++i) {
      a[i].u[0] = *(const uint4*)(arow[i] + k0 + half * 8);
      a[i].u[1] = *(const uint4*)(arow[i] + k0 + 16 + half * 8);
    }
    #pragma unroll
    for (int j = 0; j < 4; ++j) {
      b[j].u[0] = *(const uint4*)(brow[j] + k0 + half * 8);
      b[j].u[1] = *(const uint4*)(brow[j] + k0 + 16 + half * 8);
    }
    #pragma unroll
    for (int i = 0; i < 2; ++i)
      #pragma unroll
      for (int j = 0; j < 4; ++j)
        acc[i][j].v = __builtin_amdgcn_wmma_f32_16x16x32_bf16(
            false, a[i].v, false, b[j].v, (short)0, acc[i][j].v, false, false);
  }

  #pragma unroll
  for (int i = 0; i < 2; ++i) {
    #pragma unroll
    for (int j = 0; j < 4; ++j) {
      const int n = n0 + j * 16 + c16;
      const float bn = bias[n];
      #pragma unroll
      for (int r = 0; r < 8; ++r) {
        int m = m0 + i * 16 + r + half * 8;
        out[(size_t)m * DIMC + n] = acc[i][j].f[r] + bn;
      }
    }
  }
}

// ---------------------------------------------------------------------------
// Host launcher
// ---------------------------------------------------------------------------
extern "C" void kernel_launch(void* const* d_in, const int* in_sizes, int n_in,
                              void* d_out, int out_size, void* d_ws, size_t ws_size,
                              hipStream_t stream) {
  const float*         x      = (const float*)d_in[0];
  const unsigned char* mask   = (const unsigned char*)d_in[1];   // jax bool = 1 byte
  const float*         w_qkv  = (const float*)d_in[2];
  const float*         w_proj = (const float*)d_in[3];
  const float*         b_proj = (const float*)d_in[4];
  const float*         rpb    = (const float*)d_in[5];
  const int*           rel    = (const int*)d_in[6];
  const int num_rel = in_sizes[5] / NH;

  // workspace carve-out (256B aligned)
  char* ws = (char*)d_ws;
  size_t off = 0;
  auto carve = [&](size_t bytes) -> void* {
    void* p = ws + off;
    off = (off + bytes + 255) & ~(size_t)255;
    return p;
  };
  unsigned short* Xb     = (unsigned short*)carve((size_t)MTOT * DIMC * 2);
  unsigned short* Wqkvb  = (unsigned short*)carve((size_t)768 * DIMC * 2);
  unsigned short* Wprojb = (unsigned short*)carve((size_t)DIMC * DIMC * 2);
  unsigned short* Qb     = (unsigned short*)carve((size_t)BB * NH * NPAD * DH * 2);
  unsigned short* Kb     = (unsigned short*)carve((size_t)BB * NH * NPAD * DH * 2);
  unsigned short* Vtb    = (unsigned short*)carve((size_t)BB * NH * DH * NPAD * 2);
  unsigned short* AOb    = (unsigned short*)carve((size_t)MTOT * DIMC * 2);
  (void)ws_size; (void)n_in; (void)out_size;

  const int nx = MTOT * DIMC;
  cvt_f32_bf16<<<(nx + 255) / 256, 256, 0, stream>>>(x, Xb, nx);
  cvt_f32_bf16<<<(768 * DIMC + 255) / 256, 256, 0, stream>>>(w_qkv, Wqkvb, 768 * DIMC);
  cvt_f32_bf16<<<(DIMC * DIMC + 255) / 256, 256, 0, stream>>>(w_proj, Wprojb, DIMC * DIMC);

  qkv_gemm<<<dim3(MTOT / 32, 768 / 64), 32, 0, stream>>>(Xb, Wqkvb, Qb, Kb, Vtb);

  attn_kernel<<<dim3(BB * NH, NTOK / 16), 32, 0, stream>>>(
      Qb, Kb, Vtb, rpb, rel, mask, AOb, num_rel);

  proj_gemm<<<dim3(MTOT / 32, DIMC / 64), 32, 0, stream>>>(
      AOb, Wprojb, b_proj, (float*)d_out);
}
